// RelativeAttention_88364657148489
// MI455X (gfx1250) — matrix operand order repaired and checked by reading, not compiled, not run on hardware
//
#include <hip/hip_runtime.h>
#include <cstdint>
#include <cstddef>

// ---------------- problem constants (B=2, S=2048, D=1024, H=16, dk=64) -----
#define SEQ    2048
#define NHEAD  16
#define DMODEL 1024
#define DKH    64
#define NBATCH 2
#define PE_PAD 192     // zero-padded extra pe rows for skewed (masked) indices

typedef __attribute__((ext_vector_type(16))) __bf16 v16bf;
typedef __attribute__((ext_vector_type(8)))  float  v8f;
typedef __attribute__((ext_vector_type(4)))  unsigned int u32x4;
typedef __attribute__((ext_vector_type(8)))  int i32x8;
typedef __attribute__((ext_vector_type(4)))  int i32x4;

__device__ inline v8f wmma_bf16(v16bf a, v16bf b, v8f c) {
  // D(16x16,f32) = A(16x32,bf16) * B(32x16,bf16) + C
  return __builtin_amdgcn_wmma_f32_16x16x32_bf16(false, a, false, b,
                                                 (short)0, c, false, false);
}

__device__ inline v8f zero8() {
  v8f z;
#pragma unroll
  for (int i = 0; i < 8; ++i) z[i] = 0.0f;
  return z;
}

// A fragment (16x32 bf16, M x K), source is row-major [M][K] with given
// bf16 stride. ISA layout: lanes 0-15 row M=lane, VGPR0..3 = K pairs 0..7,
// VGPR4..7 = K 16..23; lanes 16-31 same rows, K 8..15 and 24..31.
__device__ inline v16bf load_frag_a(const __bf16* base, int stride) {
  const int lane = threadIdx.x & 31;
  const int m  = lane & 15;
  const int kh = lane >> 4;
  const uint32_t* row = (const uint32_t*)(base + (size_t)m * stride);
  union { v16bf v; uint32_t u[8]; } f;
  const int p0 = kh * 4;
#pragma unroll
  for (int i = 0; i < 4; ++i) {
    f.u[i]     = row[p0 + i];
    f.u[4 + i] = row[8 + p0 + i];
  }
  return f.v;
}

// B fragment (32x16 bf16, K x N) loaded from N-major storage [N][K]
// (i.e. we store B transposed so each lane reads contiguous K pairs).
// Layout matches A: lanes 0-15 VGPR0..3 = K 0..7, VGPR4..7 = K 16..23 (N=lane),
// lanes 16-31 K 8..15 and 24..31.
__device__ inline v16bf load_frag_b(const __bf16* base, int stride) {
  const int lane = threadIdx.x & 31;
  const int n  = lane & 15;
  const int kh = lane >> 4;
  const uint32_t* col = (const uint32_t*)(base + (size_t)n * stride);
  union { v16bf v; uint32_t u[8]; } f;
#pragma unroll
  for (int i = 0; i < 4; ++i) {
    f.u[i]     = col[kh * 4 + i];
    f.u[4 + i] = col[8 + kh * 4 + i];
  }
  return f.v;
}

// ---------------------------------------------------------------------------
// TDM: DMA a 64x64 bf16 tile (rows contiguous, row stride 64 elems) from
// global into LDS, inserting 1 dword of padding every 32 dwords so the LDS
// image has a 66-bf16 row stride (matches Ks[64][66]).
// D# packing per CDNA5 ISA.
// ---------------------------------------------------------------------------
__device__ inline void tdm_load_tile64x64(const __bf16* gsrc, void* lds_dst) {
  const uint64_t ga  = (uint64_t)(uintptr_t)gsrc;
  const uint32_t lds = (uint32_t)(uintptr_t)lds_dst;   // low 32 bits = LDS offset

  u32x4 g0;
  g0[0] = 1u;                                   // count=1 (valid), user desc
  g0[1] = lds;                                  // lds_addr (bytes)
  g0[2] = (uint32_t)ga;                         // global_addr[31:0]
  g0[3] = (uint32_t)(ga >> 32) | (2u << 30);    // global_addr[56:32] | type=2

  i32x8 g1;
  g1[0] = (int)((1u << 16)        // data_size = 1 -> 2 bytes
              | (1u << 20)        // pad_enable
              | (4u << 22)        // pad_interval: every 32 dwords (128 B)
              | (0u << 25));      // pad_amount: 1 dword
  g1[1] = (int)(64u << 16);       // tensor_dim0[15:0] = 64 (row length, elems)
  g1[2] = (int)(64u << 16);       // tensor_dim1[15:0] = 64 (rows)
  g1[3] = (int)(64u << 16);       // tile_dim0 = 64
  g1[4] = 64;                     // tile_dim1 = 64, tile_dim2 = 0
  g1[5] = 64;                     // tensor_dim0_stride[31:0] = 64 elems
  g1[6] = 0;
  g1[7] = 0;

  i32x4 gz;
  gz[0] = 0; gz[1] = 0; gz[2] = 0; gz[3] = 0;

#if defined(__clang_major__) && (__clang_major__ >= 23)
  i32x8 gz8;
#pragma unroll
  for (int i = 0; i < 8; ++i) gz8[i] = 0;
  __builtin_amdgcn_tensor_load_to_lds(g0, g1, gz, gz, gz8, 0);
#else
  __builtin_amdgcn_tensor_load_to_lds(g0, g1, gz, gz, 0);
#endif
}

// ---------------------------------------------------------------------------
// GEMM: out = X(fp32 [M,1024]) @ W(fp32 [1024,1024]) + bias
// MODE 0: write bf16 in [B,H,S,dk] layout (projection for Q/K/V)
// MODE 1: write fp32 row-major [M,1024] (final output projection)
// Grid: (M/128, 1024/128), 256 threads (8 waves). Each wave: 32x64 C tile.
// ---------------------------------------------------------------------------
template <int MODE>
__global__ __launch_bounds__(256) void gemm1024(const float* __restrict__ X,
                                                const float* __restrict__ W,
                                                const float* __restrict__ bias,
                                                void* __restrict__ outp) {
  __shared__ __bf16 Xs[128][66];   // [m][k]
  __shared__ __bf16 Ws[128][66];   // [n][k]  (W tile transposed)

  const int tid   = threadIdx.x;
  const int lane  = tid & 31;
  const int wv    = tid >> 5;
  const int laneq = lane & 15;
  const int kh    = lane >> 4;
  const int wm    = wv & 3;        // 4 waves along M (32 rows each)
  const int wn    = wv >> 2;       // 2 waves along N (64 cols each)
  const int m0    = blockIdx.x * 128;
  const int n0    = blockIdx.y * 128;

  v8f acc[2][4];
#pragma unroll
  for (int mi = 0; mi < 2; ++mi)
#pragma unroll
    for (int ni = 0; ni < 4; ++ni) acc[mi][ni] = zero8();

  for (int kc = 0; kc < DMODEL; kc += 64) {
    // stage X tile 128x64 (fp32 -> bf16)
#pragma unroll
    for (int ii = 0; ii < 8; ++ii) {
      int idx = tid + 256 * ii;          // float4 granules, 2048 total
      int row = idx >> 4, c4 = idx & 15;
      float4 v = *(const float4*)(X + (size_t)(m0 + row) * DMODEL + kc + c4 * 4);
      Xs[row][c4 * 4 + 0] = (__bf16)v.x;
      Xs[row][c4 * 4 + 1] = (__bf16)v.y;
      Xs[row][c4 * 4 + 2] = (__bf16)v.z;
      Xs[row][c4 * 4 + 3] = (__bf16)v.w;
    }
    // stage W tile 64x128 transposed (fp32 -> bf16)
#pragma unroll
    for (int ii = 0; ii < 8; ++ii) {
      int idx = tid + 256 * ii;
      int row = idx >> 5, c4 = idx & 31;  // row = k (0..63)
      float4 v = *(const float4*)(W + (size_t)(kc + row) * DMODEL + n0 + c4 * 4);
      Ws[c4 * 4 + 0][row] = (__bf16)v.x;
      Ws[c4 * 4 + 1][row] = (__bf16)v.y;
      Ws[c4 * 4 + 2][row] = (__bf16)v.z;
      Ws[c4 * 4 + 3][row] = (__bf16)v.w;
    }
    __syncthreads();

#pragma unroll
    for (int ks = 0; ks < 64; ks += 32) {
      v16bf a0 = load_frag_a(&Xs[wm * 32 + 0][ks], 66);
      v16bf a1 = load_frag_a(&Xs[wm * 32 + 16][ks], 66);
#pragma unroll
      for (int ni = 0; ni < 4; ++ni) {
        v16bf bf = load_frag_b(&Ws[wn * 64 + ni * 16][ks], 66);
        acc[0][ni] = wmma_bf16(a0, bf, acc[0][ni]);
        acc[1][ni] = wmma_bf16(a1, bf, acc[1][ni]);
      }
    }
    __syncthreads();
  }

  // epilogue: C layout => VGPR r: lanes 0-15 M=r, lanes 16-31 M=r+8; N=laneq
#pragma unroll
  for (int mi = 0; mi < 2; ++mi)
#pragma unroll
    for (int ni = 0; ni < 4; ++ni) {
      int ncol = n0 + wn * 64 + ni * 16 + laneq;
      float bv = bias[ncol];
#pragma unroll
      for (int r = 0; r < 8; ++r) {
        int mrow = m0 + wm * 32 + mi * 16 + r + 8 * kh;
        float v = acc[mi][ni][r] + bv;
        if (MODE == 0) {
          __bf16* o = (__bf16*)outp;
          int b = mrow >> 11, s = mrow & (SEQ - 1);
          int h = ncol >> 6, dkk = ncol & (DKH - 1);
          o[(((size_t)b * NHEAD + h) * SEQ + s) * DKH + dkk] = (__bf16)v;
        } else {
          ((float*)outp)[(size_t)mrow * DMODEL + ncol] = v;
        }
      }
    }
}

// key_pe fp32 [S,64] -> bf16 with PE_PAD zero rows appended
__global__ __launch_bounds__(256) void pe_cast(const float* __restrict__ pe,
                                               __bf16* __restrict__ out) {
  int i = blockIdx.x * 256 + threadIdx.x;
  const int total = (SEQ + PE_PAD) * DKH;
  if (i < total) out[i] = (i < SEQ * DKH) ? (__bf16)pe[i] : (__bf16)0.0f;
}

// ---------------------------------------------------------------------------
// Flash-style causal attention with relative-position (Transformer-XL skew).
// score[i,j] = (q_i.k_j + q_i.pe[S-1-(i-j)]) * (1/8),  j <= i.
// Grid: (S/128, B*H), 256 threads. Wave w owns rows iw = i0+16w .. +15.
// K tile is staged by the Tensor Data Mover (async, padded LDS image);
// V tile is staged+transposed by the VALU path in parallel.
// ---------------------------------------------------------------------------
__global__ __launch_bounds__(256) void attn_kernel(const __bf16* __restrict__ Qb,
                                                   const __bf16* __restrict__ Kb,
                                                   const __bf16* __restrict__ Vb,
                                                   const __bf16* __restrict__ pe,
                                                   float* __restrict__ Xatt) {
  __shared__ __bf16 Ks[64][66];        // K tile  [j][dk]   (TDM-written)
  __shared__ __bf16 Vt[64][66];        // V tile transposed [dk][j]
  __shared__ __bf16 Ps[8][16][64];     // per-wave P staging (C->A relayout)

  const int tid   = threadIdx.x;
  const int lane  = tid & 31;
  const int w     = tid >> 5;
  const int laneq = lane & 15;
  const int kh    = lane >> 4;
  const int bh    = blockIdx.y;               // b*H + h
  const int i0    = blockIdx.x * 128;
  const int iw    = i0 + 16 * w;

  const __bf16* Qh = Qb + (size_t)bh * SEQ * DKH;
  const __bf16* Kh = Kb + (size_t)bh * SEQ * DKH;
  const __bf16* Vh = Vb + (size_t)bh * SEQ * DKH;

  // Q fragments for this wave's 16 rows (kept resident)
  v16bf qa0 = load_frag_a(Qh + (size_t)iw * DKH + 0, DKH);
  v16bf qa1 = load_frag_a(Qh + (size_t)iw * DKH + 32, DKH);

  v8f O[4];
#pragma unroll
  for (int ot = 0; ot < 4; ++ot) O[ot] = zero8();
  float mrun[8], lrun[8];
#pragma unroll
  for (int r = 0; r < 8; ++r) { mrun[r] = -1e30f; lrun[r] = 0.0f; }

  const int ntiles = i0 / 64 + 2;   // key tiles covering j <= i0+127
  for (int t = 0; t < ntiles; ++t) {
    const int j0 = t * 64;
    // ---- K tile via Tensor Data Mover (one wave issues; EXEC ignored) ----
    if (w == 0) {
      tdm_load_tile64x64(Kh + (size_t)j0 * DKH, &Ks[0][0]);
    }
    // ---- cooperative stage of transposed V tile (overlaps the TDM) ----
    {
      const uint32_t* Vg = (const uint32_t*)(Vh + (size_t)j0 * DKH);
#pragma unroll
      for (int ii = 0; ii < 8; ++ii) {
        int idx = tid + 256 * ii;           // 2048 dwords
        int row = idx >> 5, c2 = idx & 31;  // row=j, c2=dk pair
        uint32_t pv = Vg[row * 32 + c2];
        Vt[c2 * 2 + 0][row] = __builtin_bit_cast(__bf16, (unsigned short)(pv & 0xffffu));
        Vt[c2 * 2 + 1][row] = __builtin_bit_cast(__bf16, (unsigned short)(pv >> 16));
      }
    }
    // prefetch next iteration's K/V tiles toward the caches
    if (t + 1 < ntiles) {
      __builtin_prefetch(Kh + (size_t)(j0 + 64) * DKH + tid * 16, 0, 1);
      __builtin_prefetch(Vh + (size_t)(j0 + 64) * DKH + tid * 16, 0, 1);
    }
    if (w == 0) {
      __builtin_amdgcn_s_wait_tensorcnt(0);   // TDM writes to Ks complete
    }
    __syncthreads();
    // Ks is written by the TDM (address passed only as descriptor data);
    // force the compiler to treat LDS as modified.
    asm volatile("" : : "v"(&Ks[0][0]) : "memory");

    if (j0 <= iw + 15) {   // wave-uniform: tile intersects causal region
      // ---- QK^T: 4 N-tiles x 2 K-chunks ----
      v8f sc[4];
#pragma unroll
      for (int nt = 0; nt < 4; ++nt) {
        sc[nt] = zero8();
        v16bf b0 = load_frag_b(&Ks[nt * 16][0], 66);
        v16bf b1 = load_frag_b(&Ks[nt * 16][32], 66);
        sc[nt] = wmma_bf16(qa0, b0, sc[nt]);
        sc[nt] = wmma_bf16(qa1, b1, sc[nt]);
      }
      // ---- relative-position term R = Q_tile x pe[window]^T (79 wide) ----
      v8f rf[5];
      const int lbase = SEQ - 16 - iw + j0;   // l = lbase + (15-di) + dj
#pragma unroll
      for (int nt = 0; nt < 5; ++nt) {
        rf[nt] = zero8();
        v16bf p0 = load_frag_b(pe + (size_t)(lbase + nt * 16) * DKH + 0, DKH);
        v16bf p1 = load_frag_b(pe + (size_t)(lbase + nt * 16) * DKH + 32, DKH);
        rf[nt] = wmma_bf16(qa0, p0, rf[nt]);
        rf[nt] = wmma_bf16(qa1, p1, rf[nt]);
      }
      // ---- assemble scores: skew via ds_bpermute lane rotation ----
      const float scale = 0.125f;
      float rowmax[8];
#pragma unroll
      for (int r = 0; r < 8; ++r) {
        const int di = r + 8 * kh;
        const int ig = iw + di;
        const int shift = 15 - di;
        const int srcLane = ((laneq + shift) & 15) + 16 * kh;
        const bool useHi = (laneq + shift) >= 16;
        float mx = -1e30f;
#pragma unroll
        for (int nt = 0; nt < 4; ++nt) {
          float rlo = __shfl(rf[nt][r],     srcLane, 32);
          float rhi = __shfl(rf[nt + 1][r], srcLane, 32);
          float rp  = useHi ? rhi : rlo;
          int j = j0 + nt * 16 + laneq;
          float s = (sc[nt][r] + rp) * scale;
          s = (j <= ig) ? s : -1e30f;
          sc[nt][r] = s;
          mx = fmaxf(mx, s);
        }
#pragma unroll
        for (int mmask = 1; mmask < 16; mmask <<= 1)
          mx = fmaxf(mx, __shfl_xor(mx, mmask, 32));
        rowmax[r] = mx;
      }
      // ---- online softmax update ----
#pragma unroll
      for (int r = 0; r < 8; ++r) {
        float mnew  = fmaxf(mrun[r], rowmax[r]);
        float alpha = __expf(mrun[r] - mnew);
        float rs = 0.0f;
#pragma unroll
        for (int nt = 0; nt < 4; ++nt) {
          float p = __expf(sc[nt][r] - mnew);
          sc[nt][r] = p;
          rs += p;
        }
#pragma unroll
        for (int mmask = 1; mmask < 16; mmask <<= 1)
          rs += __shfl_xor(rs, mmask, 32);
        lrun[r] = lrun[r] * alpha + rs;
        mrun[r] = mnew;
#pragma unroll
        for (int ot = 0; ot < 4; ++ot) O[ot][r] *= alpha;
      }
      // ---- P: C-layout -> A-layout via per-wave LDS staging ----
#pragma unroll
      for (int nt = 0; nt < 4; ++nt)
#pragma unroll
        for (int r = 0; r < 8; ++r)
          Ps[w][r + 8 * kh][nt * 16 + laneq] = (__bf16)sc[nt][r];
      // same-wave DS ops are in-order; no barrier needed
      v16bf pa0 = load_frag_a(&Ps[w][0][0], 64);
      v16bf pa1 = load_frag_a(&Ps[w][0][32], 64);
      // ---- O += P x V ----
#pragma unroll
      for (int ot = 0; ot < 4; ++ot) {
        v16bf v0 = load_frag_b(&Vt[ot * 16][0], 66);
        v16bf v1 = load_frag_b(&Vt[ot * 16][32], 66);
        O[ot] = wmma_bf16(pa0, v0, O[ot]);
        O[ot] = wmma_bf16(pa1, v1, O[ot]);
      }
    }
    __syncthreads();
  }

  // ---- epilogue: normalize and write fp32 attention output [B,S,D] ----
  const int b = bh >> 4;            // H = 16
  const int h = bh & (NHEAD - 1);
#pragma unroll
  for (int ot = 0; ot < 4; ++ot)
#pragma unroll
    for (int r = 0; r < 8; ++r) {
      int ig  = iw + r + 8 * kh;
      int col = h * DKH + ot * 16 + laneq;
      Xatt[((size_t)(b * SEQ + ig)) * DMODEL + col] = O[ot][r] / lrun[r];
    }
}

// ---------------------------------------------------------------------------
extern "C" void kernel_launch(void* const* d_in, const int* in_sizes, int n_in,
                              void* d_out, int out_size, void* d_ws, size_t ws_size,
                              hipStream_t stream) {
  const float* query  = (const float*)d_in[0];
  const float* key    = (const float*)d_in[1];
  const float* value  = (const float*)d_in[2];
  const float* key_pe = (const float*)d_in[3];
  const float* Wq = (const float*)d_in[4];
  const float* bq = (const float*)d_in[5];
  const float* Wk = (const float*)d_in[6];
  const float* bk = (const float*)d_in[7];
  const float* Wv = (const float*)d_in[8];
  const float* bv = (const float*)d_in[9];
  const float* Wo = (const float*)d_in[10];
  const float* bo = (const float*)d_in[11];
  (void)in_sizes; (void)n_in; (void)out_size; (void)ws_size;

  // workspace carve-out
  char* ws = (char*)d_ws;
  const size_t qkvBytes = (size_t)NBATCH * NHEAD * SEQ * DKH * 2;  // 8 MB each
  __bf16* Qb  = (__bf16*)(ws);
  __bf16* Kb  = (__bf16*)(ws + qkvBytes);
  __bf16* Vb  = (__bf16*)(ws + 2 * qkvBytes);
  __bf16* peb = (__bf16*)(ws + 3 * qkvBytes);
  float*  Xat = (float*)(ws + 3 * qkvBytes + (size_t)512 * 1024);

  const int Mrows = NBATCH * SEQ;                 // 4096
  dim3 gGemm(Mrows / 128, DMODEL / 128);          // (32, 8)

  gemm1024<0><<<gGemm, 256, 0, stream>>>(query, Wq, bq, (void*)Qb);
  gemm1024<0><<<gGemm, 256, 0, stream>>>(key,   Wk, bk, (void*)Kb);
  gemm1024<0><<<gGemm, 256, 0, stream>>>(value, Wv, bv, (void*)Vb);

  {
    int total = (SEQ + PE_PAD) * DKH;
    pe_cast<<<(total + 255) / 256, 256, 0, stream>>>(key_pe, peb);
  }

  attn_kernel<<<dim3(SEQ / 128, NBATCH * NHEAD), 256, 0, stream>>>(Qb, Kb, Vb,
                                                                   peb, Xat);

  gemm1024<1><<<gGemm, 256, 0, stream>>>(Xat, Wo, bo, d_out);
}
